// minGRUCell_15977278341676
// MI455X (gfx1250) — matrix operand order repaired and checked by reading, not compiled, not run on hardware
//
#include <hip/hip_runtime.h>
#include <hip/hip_bf16.h>

// ---------------------------------------------------------------------------
// minGRU (log-space parallel scan), MI455X / gfx1250
//   K1: fused dual GEMM (x@Wz, x@Wh), split-bf16 WMMA, async-LDS double buffer
//   K2: per-chunk scan aggregates    (64 chunks of 64 steps)
//   K3: carry scan across chunks     (seeded by log_g(h0))
//   K4: apply pass, writes exp(log_h) in place over d_out
// ---------------------------------------------------------------------------

typedef __attribute__((ext_vector_type(16))) __bf16 v16bf;
typedef __attribute__((ext_vector_type(8)))  float  v8f;
typedef __attribute__((ext_vector_type(4)))  int    v4i;

#define AS1 __attribute__((address_space(1)))
#define AS3 __attribute__((address_space(3)))

#if __has_builtin(__builtin_amdgcn_global_load_async_to_lds_b128)
#define ASYNC_LDS 1
#else
#define ASYNC_LDS 0
#endif

#if __has_builtin(__builtin_amdgcn_s_wait_asynccnt)
#define WAIT_ASYNC(n) __builtin_amdgcn_s_wait_asynccnt(n)
#else
#define WAIT_ASYNC(n) asm volatile("s_wait_asynccnt %0" ::"n"(n))
#endif

constexpr int Bb   = 8;
constexpr int Tt   = 4096;
constexpr int Dd   = 1024;
constexpr int Hh   = 1024;
constexpr int Mtot = Bb * Tt;          // 32768 rows
constexpr int BM   = 64, BN = 64, BK = 32;
constexpr int KT   = Dd / BK;          // 32 k-tiles
constexpr int NC   = 64;               // scan chunks
constexpr int CL   = Tt / NC;          // 64 steps per chunk
constexpr int XS   = 40;               // padded LDS row stride (bf16), 80B = 5*16B

union FragBF { uint4 q[2]; v16bf v; };

// ---------------- math helpers -------------------------------------------
__device__ __forceinline__ float softplusf_(float x) {
  return fmaxf(x, 0.0f) + log1pf(__expf(-fabsf(x)));
}
__device__ __forceinline__ float logaddexpf_(float a, float b) {
  float m = fmaxf(a, b);
  if (m == -__builtin_inff()) return -__builtin_inff();
  return m + log1pf(__expf(fminf(a, b) - m));
}
__device__ __forceinline__ float log_gf_(float x) {
  return (x >= 0.0f) ? __logf(x + 0.5f) : (5.0f - softplusf_(-x));
}
__device__ __forceinline__ void split2bf(float x, __bf16& hi, __bf16& lo) {
  hi = (__bf16)x;
  lo = (__bf16)(x - (float)hi);
}

// ---------------- K1: fused dual GEMM + epilogue -------------------------
// block = 256 threads = 8 waves, tile 64(M) x 64(N), waves 4x2 -> 16x32 each
__global__ __launch_bounds__(256)
void fused_dual_gemm_epilogue(const float* __restrict__ xg,
                              const float* __restrict__ Wz,
                              const float* __restrict__ bz,
                              const float* __restrict__ Wh,
                              const float* __restrict__ bh,
                              float* __restrict__ wsA,   // log_coeffs  [M,H]
                              float* __restrict__ outV)  // log_values  [M,H] (d_out reused)
{
  // split bf16 operand planes consumed by WMMA fragments
  __shared__ __align__(16) __bf16 sXhi[BM][XS], sXlo[BM][XS];   // x tile (row-major)
  __shared__ __align__(16) __bf16 sZhi[BN][XS], sZlo[BN][XS];   // Wz tile, transposed [n][k]
  __shared__ __align__(16) __bf16 sHhi[BN][XS], sHlo[BN][XS];   // Wh tile, transposed [n][k]
#if ASYNC_LDS
  // raw fp32 double buffers filled by GLOBAL_LOAD_ASYNC_TO_LDS_B128
  __shared__ __align__(16) float rawX[2][BM * BK];
  __shared__ __align__(16) float rawZ[2][BK * BN];
  __shared__ __align__(16) float rawH[2][BK * BN];
#endif

  const int tid  = threadIdx.x;
  const int lane = tid & 31;
  const int wave = tid >> 5;
  const int wm   = wave & 3;       // 0..3 -> M sub-tile
  const int wn   = wave >> 2;      // 0..1 -> N sub-tile
  const int l15  = lane & 15;
  const int lh   = lane >> 4;      // lane half
  const int m0   = blockIdx.x * BM;
  const int n0   = blockIdx.y * BN;

  v8f accZ0 = {}, accZ1 = {}, accH0 = {}, accH1 = {};

#if ASYNC_LDS
  // ---- async global->LDS tile issue (6 b128 per thread per k-tile) ----
  auto issue_tile = [&](int k0, float* rX, float* rZ, float* rH) {
    #pragma unroll
    for (int it = 0; it < 2; ++it) {
      const int j = tid + it * 256;                 // float4 granule id
      {
        const int r = j >> 3, c = (j & 7) << 2;     // x: 64 x 32
        const float* g = xg + (size_t)(m0 + r) * Dd + k0 + c;
        __builtin_amdgcn_global_load_async_to_lds_b128(
            (AS1 v4i*)g, (AS3 v4i*)(rX + j * 4), 0, 0);
      }
      {
        const int r = j >> 4, c = (j & 15) << 2;    // W: 32 x 64
        const float* gz = Wz + (size_t)(k0 + r) * Hh + n0 + c;
        const float* gh = Wh + (size_t)(k0 + r) * Hh + n0 + c;
        __builtin_amdgcn_global_load_async_to_lds_b128(
            (AS1 v4i*)gz, (AS3 v4i*)(rZ + j * 4), 0, 0);
        __builtin_amdgcn_global_load_async_to_lds_b128(
            (AS1 v4i*)gh, (AS3 v4i*)(rH + j * 4), 0, 0);
      }
    }
  };
  issue_tile(0, rawX[0], rawZ[0], rawH[0]);         // preload tile 0
#endif

  for (int kt = 0; kt < KT; ++kt) {
    const int k0 = kt * BK;

#if ASYNC_LDS
    const int cur = kt & 1;
    if (kt + 1 < KT) {
      issue_tile(k0 + BK, rawX[cur ^ 1], rawZ[cur ^ 1], rawH[cur ^ 1]);
      WAIT_ASYNC(6);      // in-order: first 6 (tile kt) done, next 6 in flight
    } else {
      WAIT_ASYNC(0);
    }
    __syncthreads();      // tile-kt raw data visible to all waves; frag reads of
                          // previous iteration's planes are complete
    // ---- convert raw fp32 -> split bf16 hi/lo planes ----
    #pragma unroll
    for (int it = 0; it < 2; ++it) {
      const int j = tid + it * 256;
      {
        const int r = j >> 3, c = (j & 7) << 2;
        const float4 xv = *(const float4*)(rawX[cur] + j * 4);
        const float f[4] = {xv.x, xv.y, xv.z, xv.w};
        #pragma unroll
        for (int e = 0; e < 4; ++e) split2bf(f[e], sXhi[r][c + e], sXlo[r][c + e]);
      }
      {
        const int r = j >> 4, c = (j & 15) << 2;
        const float4 zv = *(const float4*)(rawZ[cur] + j * 4);
        const float4 hv = *(const float4*)(rawH[cur] + j * 4);
        const float fz[4] = {zv.x, zv.y, zv.z, zv.w};
        const float fh[4] = {hv.x, hv.y, hv.z, hv.w};
        #pragma unroll
        for (int e = 0; e < 4; ++e) {
          split2bf(fz[e], sZhi[c + e][r], sZlo[c + e][r]);
          split2bf(fh[e], sHhi[c + e][r], sHlo[c + e][r]);
        }
      }
    }
    __syncthreads();
#else
    __syncthreads();   // protect previous iteration's fragment reads
    #pragma unroll
    for (int it = 0; it < 2; ++it) {
      const int j = tid + it * 256;
      {
        const int r = j >> 3, c = (j & 7) << 2;
        const float4 xv = *(const float4*)(xg + (size_t)(m0 + r) * Dd + k0 + c);
        const float f[4] = {xv.x, xv.y, xv.z, xv.w};
        #pragma unroll
        for (int e = 0; e < 4; ++e) split2bf(f[e], sXhi[r][c + e], sXlo[r][c + e]);
        if (kt + 1 < KT)
          __builtin_prefetch(xg + (size_t)(m0 + r) * Dd + k0 + BK + c, 0, 0);
      }
      {
        const int r = j >> 4, c = (j & 15) << 2;
        const float4 zv = *(const float4*)(Wz + (size_t)(k0 + r) * Hh + n0 + c);
        const float4 hv = *(const float4*)(Wh + (size_t)(k0 + r) * Hh + n0 + c);
        const float fz[4] = {zv.x, zv.y, zv.z, zv.w};
        const float fh[4] = {hv.x, hv.y, hv.z, hv.w};
        #pragma unroll
        for (int e = 0; e < 4; ++e) {
          split2bf(fz[e], sZhi[c + e][r], sZlo[c + e][r]);
          split2bf(fh[e], sHhi[c + e][r], sHlo[c + e][r]);
        }
        if (kt + 1 < KT) {
          __builtin_prefetch(Wz + (size_t)(k0 + BK + r) * Hh + n0 + c, 0, 0);
          __builtin_prefetch(Wh + (size_t)(k0 + BK + r) * Hh + n0 + c, 0, 0);
        }
      }
    }
    __syncthreads();
#endif

    // ---- A fragments (16x32 bf16, ISA layout: lanes 0-15 K0-7/16-23,
    //      lanes 16-31 K8-15/24-31, 2 packed values per VGPR) ----
    const int am   = wm * 16 + l15;
    const int aklo = lh * 8;
    FragBF axh, axl;
    axh.q[0] = *(const uint4*)&sXhi[am][aklo];
    axh.q[1] = *(const uint4*)&sXhi[am][16 + aklo];
    axl.q[0] = *(const uint4*)&sXlo[am][aklo];
    axl.q[1] = *(const uint4*)&sXlo[am][16 + aklo];

    // ---- B fragments (32x16 bf16: lane n = lane&15, K 0-7/16-23 or 8-15/24-31 by lane>>4) ----
    const int bkb = lh * 8;
    #pragma unroll
    for (int sub = 0; sub < 2; ++sub) {
      const int bn = wn * 32 + sub * 16 + l15;
      FragBF bzh, bzl, bhh, bhl;
      bzh.q[0] = *(const uint4*)&sZhi[bn][bkb];
      bzh.q[1] = *(const uint4*)&sZhi[bn][bkb + 16];
      bzl.q[0] = *(const uint4*)&sZlo[bn][bkb];
      bzl.q[1] = *(const uint4*)&sZlo[bn][bkb + 16];
      bhh.q[0] = *(const uint4*)&sHhi[bn][bkb];
      bhh.q[1] = *(const uint4*)&sHhi[bn][bkb + 16];
      bhl.q[0] = *(const uint4*)&sHlo[bn][bkb];
      bhl.q[1] = *(const uint4*)&sHlo[bn][bkb + 16];

      v8f az = sub ? accZ1 : accZ0;
      v8f ah = sub ? accH1 : accH0;
      // split-precision product: hi*hi + hi*lo + lo*hi  (~fp32 accuracy)
      az = __builtin_amdgcn_wmma_f32_16x16x32_bf16(false, axh.v, false, bzh.v, (short)0, az, false, false);
      az = __builtin_amdgcn_wmma_f32_16x16x32_bf16(false, axh.v, false, bzl.v, (short)0, az, false, false);
      az = __builtin_amdgcn_wmma_f32_16x16x32_bf16(false, axl.v, false, bzh.v, (short)0, az, false, false);
      ah = __builtin_amdgcn_wmma_f32_16x16x32_bf16(false, axh.v, false, bhh.v, (short)0, ah, false, false);
      ah = __builtin_amdgcn_wmma_f32_16x16x32_bf16(false, axh.v, false, bhl.v, (short)0, ah, false, false);
      ah = __builtin_amdgcn_wmma_f32_16x16x32_bf16(false, axl.v, false, bhh.v, (short)0, ah, false, false);
      if (sub) { accZ1 = az; accH1 = ah; } else { accZ0 = az; accH0 = ah; }
    }
  }

  // ---- epilogue: k -> (log_coeffs, log_values) ----
  // C/D layout: VGPR r, lanes 0-15: (M=r, N=lane); lanes 16-31: (M=r+8, N=lane-16)
  #pragma unroll
  for (int sub = 0; sub < 2; ++sub) {
    const int n   = n0 + wn * 32 + sub * 16 + l15;
    const float bzv = bz[n];
    const float bhv = bh[n];
    const v8f az = sub ? accZ1 : accZ0;
    const v8f ah = sub ? accH1 : accH0;
    #pragma unroll
    for (int r = 0; r < 8; ++r) {
      const int   m  = m0 + wm * 16 + lh * 8 + r;
      const float kz = az[r] + bzv;
      const float kh = ah[r] + bhv;
      const float lz = -softplusf_(-kz);      // log sigmoid(k)
      const float lc = -softplusf_(kz);       // log(1 - sigmoid(k))
      const float v  = lz + log_gf_(kh);      // log_values
      const size_t idx = (size_t)m * Hh + n;
      wsA[idx]  = lc;
      outV[idx] = v;
    }
  }
}

// ---------------- K2: per-chunk aggregates --------------------------------
// operator: state' = (Acur + a, logaddexp(a + Vcur, v)), identity (0, -inf)
__global__ __launch_bounds__(256)
void scan_chunk(const float* __restrict__ A, const float* __restrict__ V,
                float* __restrict__ Asum, float* __restrict__ Vagg)
{
  const int h = blockIdx.x * 256 + threadIdx.x;
  const int c = blockIdx.y;
  const int b = blockIdx.z;
  size_t base = ((size_t)b * Tt + (size_t)c * CL) * Hh + h;
  float As = 0.0f, Vs = -__builtin_inff();
  for (int i = 0; i < CL; ++i) {
    const float a = A[base];
    const float v = V[base];
    Vs = logaddexpf_(a + Vs, v);
    As += a;
    base += Hh;
  }
  const size_t o = ((size_t)b * NC + c) * Hh + h;
  Asum[o] = As;
  Vagg[o] = Vs;
}

// ---------------- K3: carry scan across chunks ----------------------------
__global__ __launch_bounds__(256)
void scan_carry(const float* __restrict__ h0,
                const float* __restrict__ Asum, const float* __restrict__ Vagg,
                float* __restrict__ carry)
{
  const int h = blockIdx.x * 256 + threadIdx.x;
  const int b = blockIdx.y;
  float st = log_gf_(h0[(size_t)b * Hh + h]);         // log_h before first step
  carry[((size_t)b * NC) * Hh + h] = st;
  for (int c = 0; c < NC - 1; ++c) {
    const size_t o = ((size_t)b * NC + c) * Hh + h;
    st = logaddexpf_(Asum[o] + st, Vagg[o]);
    carry[o + Hh] = st;
  }
}

// ---------------- K4: apply pass, write exp(log_h) in place ---------------
__global__ __launch_bounds__(256)
void scan_apply(const float* __restrict__ A, const float* __restrict__ carry,
                float* __restrict__ V /* in: log_values, out: h */)
{
  const int h = blockIdx.x * 256 + threadIdx.x;
  const int c = blockIdx.y;
  const int b = blockIdx.z;
  float st = carry[((size_t)b * NC + c) * Hh + h];
  size_t idx = ((size_t)b * Tt + (size_t)c * CL) * Hh + h;
  for (int i = 0; i < CL; ++i) {
    st = logaddexpf_(A[idx] + st, V[idx]);
    V[idx] = __expf(st);
    idx += Hh;
  }
}

// ---------------- launch ---------------------------------------------------
extern "C" void kernel_launch(void* const* d_in, const int* in_sizes, int n_in,
                              void* d_out, int out_size, void* d_ws, size_t ws_size,
                              hipStream_t stream) {
  const float* x  = (const float*)d_in[0];
  const float* h0 = (const float*)d_in[1];
  const float* Wz = (const float*)d_in[2];
  const float* bz = (const float*)d_in[3];
  const float* Wh = (const float*)d_in[4];
  const float* bh = (const float*)d_in[5];
  float* out = (float*)d_out;

  float* wsA  = (float*)d_ws;                       // log_coeffs [M,H] (134 MB)
  float* Asum = wsA  + (size_t)Mtot * Hh;           // [B,NC,H] (2 MB)
  float* Vagg = Asum + (size_t)Bb * NC * Hh;        // [B,NC,H]
  float* cry  = Vagg + (size_t)Bb * NC * Hh;        // [B,NC,H]

  dim3 gA(Mtot / BM, Hh / BN);                      // 512 x 16 blocks
  fused_dual_gemm_epilogue<<<gA, 256, 0, stream>>>(x, Wz, bz, Wh, bh, wsA, out);

  dim3 g1(Hh / 256, NC, Bb);
  scan_chunk<<<g1, 256, 0, stream>>>(wsA, out, Asum, Vagg);

  dim3 g2(Hh / 256, Bb);
  scan_carry<<<g2, 256, 0, stream>>>(h0, Asum, Vagg, cry);

  scan_apply<<<g1, 256, 0, stream>>>(wsA, cry, out);
}
